// GMLP_2_83270825935019
// MI455X (gfx1250) — matrix operand order repaired and checked by reading, not compiled, not run on hardware
//
#include <hip/hip_runtime.h>

// ---------------- problem constants ----------------
#define B_      16
#define L_      512
#define C_      321
#define DMODEL  512
#define DINNER  1024
#define DSTATE  16
#define DTRANK  32
#define HORIZON 96
#define NBLOCKS 3
#define M_REAL  (B_ * C_)   // 5136
#define M_PAD   5248        // 41 * 128

typedef __attribute__((ext_vector_type(16))) _Float16 v16h;
typedef __attribute__((ext_vector_type(8)))  _Float16 v8h;
typedef __attribute__((ext_vector_type(8)))  float    v8f;
typedef __attribute__((ext_vector_type(4)))  int      v4i;

#if defined(__has_builtin)
#if __has_builtin(__builtin_amdgcn_global_load_async_to_lds_b128) && \
    __has_builtin(__builtin_amdgcn_s_wait_asynccnt)
#define HAVE_ASYNC_LDS 1
#endif
#endif
#ifndef HAVE_ASYNC_LDS
#define HAVE_ASYNC_LDS 0
#endif

__device__ __forceinline__ unsigned short f2h(float f) {
  return __builtin_bit_cast(unsigned short, (_Float16)f);
}
__device__ __forceinline__ float h2f(unsigned short h) {
  return (float)__builtin_bit_cast(_Float16, h);
}

#if HAVE_ASYNC_LDS
__device__ __forceinline__ void async_cp16(void* ldsDst, const void* gSrc) {
  __builtin_amdgcn_global_load_async_to_lds_b128(
      (__attribute__((address_space(1))) v4i*)gSrc,
      (__attribute__((address_space(3))) v4i*)ldsDst,
      0, 0);
}
#endif

// ---------------- weight prep: f32 -> f16 (+padding), A = -exp(A_log) -------
__global__ void prep_weights(const float* __restrict__ W_in,
                             const float* __restrict__ W_xp,
                             const float* __restrict__ b_xp,
                             const float* __restrict__ W_dt,
                             const float* __restrict__ A_log,
                             const float* __restrict__ W_out,
                             unsigned short* __restrict__ W_in_h,
                             unsigned short* __restrict__ Wxp_h,   // [128,1024] zero padded
                             float* __restrict__ bxp_pad,          // [128]
                             unsigned short* __restrict__ W_dt_h,
                             float* __restrict__ A32,
                             unsigned short* __restrict__ W_out_h)
{
  const int stride = gridDim.x * blockDim.x;
  const int tid = blockIdx.x * blockDim.x + threadIdx.x;
  for (int i = tid; i < 2 * DINNER * DMODEL; i += stride) W_in_h[i] = f2h(W_in[i]);
  for (int i = tid; i < 128 * DINNER; i += stride)
    Wxp_h[i] = f2h(i < (DTRANK + 2 * DSTATE) * DINNER ? W_xp[i] : 0.f);
  for (int i = tid; i < 128; i += stride)
    bxp_pad[i] = (i < DTRANK + 2 * DSTATE) ? b_xp[i] : 0.f;
  for (int i = tid; i < DINNER * DTRANK; i += stride) W_dt_h[i] = f2h(W_dt[i]);
  for (int i = tid; i < DINNER * DSTATE; i += stride) A32[i] = -__expf(A_log[i]);
  for (int i = tid; i < DMODEL * DINNER; i += stride) W_out_h[i] = f2h(W_out[i]);
}

// ---------------- h0 = (x - x[:, -1, :]) transposed -> f16 ------------------
__global__ void prep_h0(const float* __restrict__ x, unsigned short* __restrict__ hh)
{
  int idx = blockIdx.x * blockDim.x + threadIdx.x;
  if (idx >= M_REAL * DMODEL) return;
  int row = idx >> 9;            // b*321 + c
  int l   = idx & (DMODEL - 1);
  int b = row / C_;
  int c = row % C_;
  float v = x[((size_t)b * L_ + l) * C_ + c] - x[((size_t)b * L_ + (L_ - 1)) * C_ + c];
  hh[idx] = f2h(v);
}

// ---------------- WMMA GEMM: C = act(A[M,K] * Bw[N,K]^T + bias) -------------
// 128x128 block tile, 8 waves (2x4), each wave = 64x32 -> 4x2 16x16 WMMA tiles.
// LDS row pitch 80B (64B data + 16B pad) -> conflict-free b128 fragment reads.
// Staging: async global->LDS copies, double buffered (pipelined with WMMA).
#define LDS_PITCH 80
#define TILE_BYTES (128 * LDS_PITCH)   // 10240 per tile

template<int ACT>   // 0 = none, 1 = exact GELU, 2 = softplus
__global__ __launch_bounds__(256)
void wmma_gemm(const unsigned short* __restrict__ A,   // [M_PAD, K] f16
               const unsigned short* __restrict__ Bw,  // [N, K]     f16
               const float* __restrict__ bias,         // [N]
               float* __restrict__ Cf, int ldc,        // optional f32 out
               unsigned short* __restrict__ Ch, int ldh, int nh,   // optional f16 out
               int N, int K)
{
#if HAVE_ASYNC_LDS
  __shared__ unsigned char lds[4 * TILE_BYTES];        // {A0,B0,A1,B1}
#else
  __shared__ unsigned char lds[2 * TILE_BYTES];        // {A,B}
#endif

  const int m0   = blockIdx.y * 128;
  const int n0   = blockIdx.x * 128;
  const int tid  = threadIdx.x;
  const int lane = tid & 31;
  const int wave = tid >> 5;
  const int wm   = wave >> 2;   // 0..1
  const int wn   = wave & 3;    // 0..3
  const int lr   = lane & 15;
  const int lh   = lane >> 4;

  v8f acc[4][2];
  #pragma unroll
  for (int mi = 0; mi < 4; ++mi)
    #pragma unroll
    for (int ni = 0; ni < 2; ++ni)
      #pragma unroll
      for (int e = 0; e < 8; ++e) acc[mi][ni][e] = 0.f;

  const int aOff = lh * 16;   // A frag: lanes16-31 start at K=8 (16B)
  const int bOff = lh * 16;   // B frag: lanes16-31 start at K=8 (16B)
  const int srow = tid >> 1;         // staging: this thread's tile row (2 threads/row)
  const int scc  = (tid & 1) * 2;    // first of two 16B chunks in the 64B row

  // fragment compute from one LDS tile pair
  auto computeTile = [&](const unsigned char* tA, const unsigned char* tB) {
    v16h fa[4], fb[2];
    #pragma unroll
    for (int mi = 0; mi < 4; ++mi) {
      const unsigned char* p = tA + (wm * 64 + mi * 16 + lr) * LDS_PITCH + aOff;
      v8h lo = *(const v8h*)p;
      v8h hi = *(const v8h*)(p + 32);
      fa[mi] = __builtin_shufflevector(lo, hi, 0,1,2,3,4,5,6,7,8,9,10,11,12,13,14,15);
    }
    #pragma unroll
    for (int ni = 0; ni < 2; ++ni) {
      const unsigned char* p = tB + (wn * 32 + ni * 16 + lr) * LDS_PITCH + bOff;
      v8h lo = *(const v8h*)p;
      v8h hi = *(const v8h*)(p + 32);
      fb[ni] = __builtin_shufflevector(lo, hi, 0,1,2,3,4,5,6,7,8,9,10,11,12,13,14,15);
    }
    #pragma unroll
    for (int mi = 0; mi < 4; ++mi)
      #pragma unroll
      for (int ni = 0; ni < 2; ++ni)
        acc[mi][ni] = __builtin_amdgcn_wmma_f32_16x16x32_f16(
            false, fa[mi], false, fb[ni], (short)0, acc[mi][ni], false, false);
  };

#if HAVE_ASYNC_LDS
  // issue 4 async 16B copies (2 for A-tile, 2 for B-tile) for K-step k0 into buf
  auto issueTile = [&](int k0, int buf) {
    unsigned char* dA = lds + buf * 2 * TILE_BYTES;
    unsigned char* dB = dA + TILE_BYTES;
    #pragma unroll
    for (int j = 0; j < 2; ++j) {
      int cc = scc + j;
      async_cp16(dA + srow * LDS_PITCH + cc * 16,
                 A  + (size_t)(m0 + srow) * K + k0 + cc * 8);
      async_cp16(dB + srow * LDS_PITCH + cc * 16,
                 Bw + (size_t)(n0 + srow) * K + k0 + cc * 8);
    }
  };

  const int nk = K >> 5;
  issueTile(0, 0);
  for (int kb = 0; kb < nk - 1; ++kb) {
    const int p = kb & 1;
    issueTile((kb + 1) << 5, 1 - p);            // prefetch next tile
    __builtin_amdgcn_s_wait_asynccnt(4);        // tile kb landed; next still in flight
    __syncthreads();
    computeTile(lds + p * 2 * TILE_BYTES, lds + p * 2 * TILE_BYTES + TILE_BYTES);
    __syncthreads();
  }
  {
    const int p = (nk - 1) & 1;
    __builtin_amdgcn_s_wait_asynccnt(0);
    __syncthreads();
    computeTile(lds + p * 2 * TILE_BYTES, lds + p * 2 * TILE_BYTES + TILE_BYTES);
  }
#else
  for (int k0 = 0; k0 < K; k0 += 32) {
    #pragma unroll
    for (int j = 0; j < 2; ++j) {
      int cc = scc + j;
      uint4 da = *(const uint4*)(A  + (size_t)(m0 + srow) * K + k0 + cc * 8);
      uint4 db = *(const uint4*)(Bw + (size_t)(n0 + srow) * K + k0 + cc * 8);
      *(uint4*)(lds + srow * LDS_PITCH + cc * 16) = da;
      *(uint4*)(lds + TILE_BYTES + srow * LDS_PITCH + cc * 16) = db;
    }
    __syncthreads();
    computeTile(lds, lds + TILE_BYTES);
    __syncthreads();
  }
#endif

  // epilogue: C/D layout -> lane holds col (lane&15), rows v + 8*(lane>>4)
  const int colBase = n0 + wn * 32;
  const int rowBase = m0 + wm * 64;
  #pragma unroll
  for (int mi = 0; mi < 4; ++mi) {
    #pragma unroll
    for (int ni = 0; ni < 2; ++ni) {
      int col = colBase + ni * 16 + lr;
      int r0  = rowBase + mi * 16 + (lh << 3);
      float bv = bias[col];
      #pragma unroll
      for (int v = 0; v < 8; ++v) {
        float xv = acc[mi][ni][v] + bv;
        if (ACT == 1) xv = 0.5f * xv * (1.f + erff(xv * 0.70710678118654752f));
        else if (ACT == 2) xv = fmaxf(xv, 0.f) + log1pf(__expf(-fabsf(xv)));
        int row = r0 + v;
        if (Cf)  Cf[(size_t)row * ldc + col] = xv;
        if (Ch && col < nh) Ch[(size_t)row * ldh + col] = f2h(xv);
      }
    }
  }
}

// ---------------- selective-scan: one thread per (b, d), 16 states ----------
__global__ __launch_bounds__(256)
void ssm_scan(const float* __restrict__ delta,  // [M_PAD, 1024]
              const float* __restrict__ xz,     // [M_PAD, 2048] (x_ | z)
              const float* __restrict__ dbc,    // [M_PAD, 128]  cols 32..47=B, 48..63=C
              const float* __restrict__ A32,    // [1024, 16]  (= -exp(A_log))
              const float* __restrict__ Dp,     // [1024]
              unsigned short* __restrict__ yz)  // [M_PAD, 1024] f16 gated output
{
  int tid = blockIdx.x * blockDim.x + threadIdx.x;   // 0..16383
  int b = tid >> 10;
  int d = tid & (DINNER - 1);
  float ar[DSTATE], h[DSTATE];
  #pragma unroll
  for (int n = 0; n < DSTATE; ++n) { ar[n] = A32[d * DSTATE + n]; h[n] = 0.f; }
  float Dpd = Dp[d];

  for (int s = 0; s < C_; ++s) {
    size_t row = (size_t)b * C_ + s;
    float dlt = delta[row * DINNER + d];
    float xv  = xz[row * 2048 + d];
    float zv  = xz[row * 2048 + DINNER + d];
    const float* bc = dbc + row * 128;
    float dx = dlt * xv;
    float accv = 0.f;
    #pragma unroll
    for (int n = 0; n < DSTATE; ++n) {
      float a = __expf(dlt * ar[n]);          // exp(delta * A) in (0,1]
      h[n] = fmaf(a, h[n], dx * bc[DTRANK + n]);
      accv = fmaf(h[n], bc[DTRANK + DSTATE + n], accv);
    }
    float y = accv + Dpd * xv;
    yz[row * DINNER + d] = f2h(y * zv);
  }
}

// ---------------- final head: out[b,t,c] = h@W_op^T + b_op + seq_last -------
__global__ void final_proj(const unsigned short* __restrict__ hh,
                           const float* __restrict__ W_op,
                           const float* __restrict__ b_op,
                           const float* __restrict__ x,
                           float* __restrict__ out)
{
  int t   = blockIdx.x;                                   // 0..95
  int row = blockIdx.y * blockDim.x + threadIdx.x;        // b*321 + c
  if (row >= M_REAL) return;
  int b = row / C_;
  int c = row % C_;
  float acc = b_op[t];
  const float* w = W_op + (size_t)t * DMODEL;
  const unsigned short* h = hh + (size_t)row * DMODEL;
  #pragma unroll 4
  for (int l = 0; l < DMODEL; ++l) acc = fmaf(h2f(h[l]), w[l], acc);
  float last = x[((size_t)b * L_ + (L_ - 1)) * C_ + c];
  out[((size_t)b * HORIZON + t) * C_ + c] = acc + last;
}

// ---------------- host launch ----------------
extern "C" void kernel_launch(void* const* d_in, const int* in_sizes, int n_in,
                              void* d_out, int out_size, void* d_ws, size_t ws_size,
                              hipStream_t stream)
{
  const float* x     = (const float*)d_in[0];
  const float* W_in  = (const float*)d_in[1];
  const float* b_in  = (const float*)d_in[2];
  const float* W_xp  = (const float*)d_in[3];
  const float* b_xp  = (const float*)d_in[4];
  const float* W_dt  = (const float*)d_in[5];
  const float* b_dt  = (const float*)d_in[6];
  const float* A_log = (const float*)d_in[7];
  const float* Dp    = (const float*)d_in[8];
  const float* W_out = (const float*)d_in[9];
  const float* b_out = (const float*)d_in[10];
  const float* W_op  = (const float*)d_in[11];
  const float* b_op  = (const float*)d_in[12];
  float* out = (float*)d_out;

  // carve workspace (deterministic layout every call)
  char* p = (char*)d_ws;
  auto carve = [&](size_t bytes) -> char* {
    char* r = p;
    p += (bytes + 255) & ~(size_t)255;
    return r;
  };
  unsigned short* hh      = (unsigned short*)carve((size_t)M_PAD * DMODEL * 2);
  unsigned short* x_h     = (unsigned short*)carve((size_t)M_PAD * DINNER * 2);
  unsigned short* dbc_h   = (unsigned short*)carve((size_t)M_PAD * DTRANK * 2);
  unsigned short* yz_h    = (unsigned short*)carve((size_t)M_PAD * DINNER * 2);
  float*          xz_f    = (float*)carve((size_t)M_PAD * 2048 * 4);
  float*          dbc_f   = (float*)carve((size_t)M_PAD * 128 * 4);
  float*          delta_f = (float*)carve((size_t)M_PAD * DINNER * 4);
  unsigned short* W_in_h  = (unsigned short*)carve((size_t)2 * DINNER * DMODEL * 2);
  unsigned short* Wxp_h   = (unsigned short*)carve((size_t)128 * DINNER * 2);
  unsigned short* W_dt_h  = (unsigned short*)carve((size_t)DINNER * DTRANK * 2);
  unsigned short* W_out_h = (unsigned short*)carve((size_t)DMODEL * DINNER * 2);
  float*          A32     = (float*)carve((size_t)DINNER * DSTATE * 4);
  float*          bxp_pad = (float*)carve(128 * 4);

  prep_weights<<<512, 256, 0, stream>>>(W_in, W_xp, b_xp, W_dt, A_log, W_out,
                                        W_in_h, Wxp_h, bxp_pad, W_dt_h, A32, W_out_h);
  prep_h0<<<(M_REAL * DMODEL + 255) / 256, 256, 0, stream>>>(x, hh);

  const int gy = M_PAD / 128;  // 41
  for (int blk = 0; blk < NBLOCKS; ++blk) {
    // xz = h @ W_in^T + b_in       [M,2048], also f16 copy of x_ (cols 0..1023)
    wmma_gemm<0><<<dim3(2 * DINNER / 128, gy), 256, 0, stream>>>(
        hh, W_in_h, b_in, xz_f, 2 * DINNER, x_h, DINNER, DINNER, 2 * DINNER, DMODEL);
    // dbc = x_ @ W_xp^T + b_xp     [M,128 padded], f16 copy of delta_raw (cols 0..31)
    wmma_gemm<0><<<dim3(1, gy), 256, 0, stream>>>(
        x_h, Wxp_h, bxp_pad, dbc_f, 128, dbc_h, DTRANK, DTRANK, 128, DINNER);
    // delta = softplus(delta_raw @ W_dt^T + b_dt)   [M,1024]
    wmma_gemm<2><<<dim3(DINNER / 128, gy), 256, 0, stream>>>(
        dbc_h, W_dt_h, b_dt, delta_f, DINNER, nullptr, 0, 0, DINNER, DTRANK);
    // selective scan + gate -> yz (f16)
    ssm_scan<<<(B_ * DINNER) / 256, 256, 0, stream>>>(delta_f, xz_f, dbc_f, A32, Dp, yz_h);
    // h = gelu((y*z) @ W_out^T + b_out)   -> next block input (f16)
    wmma_gemm<1><<<dim3(DMODEL / 128, gy), 256, 0, stream>>>(
        yz_h, W_out_h, b_out, nullptr, 0, hh, DMODEL, DMODEL, DMODEL, DINNER);
  }

  final_proj<<<dim3(HORIZON, (M_REAL + 255) / 256), 256, 0, stream>>>(hh, W_op, b_op, x, out);
}
